// BasicSubGraphLearner_24335284699139
// MI455X (gfx1250) — compile-verified, matrix operand order checked by reading, not run on hardware
//
#include <hip/hip_runtime.h>

#define N_NODES 8192
#define D_FEAT  256
#define P_PERS  8
#define K_DIM   (P_PERS * D_FEAT)   // 2048
#define EPS_THR 0.5f

typedef __attribute__((ext_vector_type(16))) _Float16 v16h;
typedef __attribute__((ext_vector_type(8)))  _Float16 v8h;
typedef __attribute__((ext_vector_type(8)))  float    v8f;

// ---------------------------------------------------------------------------
// Kernel 1: scatter edge list into a dense N x N bitmask (1 bit per entry).
// ---------------------------------------------------------------------------
__global__ void edge_mask_kernel(const int* __restrict__ ei, int E,
                                 unsigned* __restrict__ mask) {
    int e = blockIdx.x * blockDim.x + threadIdx.x;
    if (e >= E) return;
    int r = ei[e];          // full_edge_index[0][e]
    int c = ei[E + e];      // full_edge_index[1][e]
    size_t bitIdx = (size_t)r * N_NODES + c;
    atomicOr(&mask[bitIdx >> 5], 1u << (c & 31));
}

// ---------------------------------------------------------------------------
// Kernel 2: ctx[n, p*256+d] = f16( (w[p,d]*x[n,d]) / max(||w[p,:]*x[n,:]||, 1e-12) )
// One 256-thread block per (p, n).
// ---------------------------------------------------------------------------
__global__ void ctx_normalize_kernel(const float* __restrict__ x,
                                     const float* __restrict__ w,
                                     _Float16* __restrict__ ctx) {
    int p = blockIdx.x;
    int n = blockIdx.y;
    int d = threadIdx.x;
    float y = w[p * D_FEAT + d] * x[n * D_FEAT + d];

    __shared__ float s[D_FEAT];
    s[d] = y * y;
    __syncthreads();
    for (int o = D_FEAT / 2; o > 0; o >>= 1) {
        if (d < o) s[d] += s[d + o];
        __syncthreads();
    }
    float scale = 1.0f / fmaxf(sqrtf(s[0]), 1e-12f);
    ctx[(size_t)n * K_DIM + p * D_FEAT + d] = (_Float16)(y * scale);
}

// ---------------------------------------------------------------------------
// Kernel 3: adj = (1/P) * Ctx @ Ctx^T with fused mask/epsilon/self-loop
// epilogue. One wave (32 threads) computes a 64x64 output tile as a 4x4 grid
// of v_wmma_f32_16x16x32_f16 tiles, K-loop over 2048 in steps of 32.
//
// A-fragment per lane: row = tileRow + (lane&15); two contiguous v8h loads at
// k0+koff and k0+16+koff, koff = 8*(lane>=16)  (matches the 16-bit 16x32 A
// layout in CDNA5 ISA). B-fragment for Ctx^T uses the identical
// loader (lane = output column).
// ---------------------------------------------------------------------------
__global__ __launch_bounds__(32) void adj_wmma_kernel(
    const _Float16* __restrict__ ctx,
    const unsigned* __restrict__ mask,
    float* __restrict__ out) {

    const int tm   = blockIdx.y * 64;
    const int tn   = blockIdx.x * 64;
    const int lane = threadIdx.x;
    const int r    = lane & 15;
    const int koff = (lane >> 4) << 3;   // 0 or 8

    v8f acc[4][4] = {};

    const _Float16* A0 = ctx + (size_t)(tm + r) * K_DIM;
    const _Float16* B0 = ctx + (size_t)(tn + r) * K_DIM;

    for (int k0 = 0; k0 < K_DIM; k0 += 32) {
        v16h a[4], b[4];
        #pragma unroll
        for (int i = 0; i < 4; ++i) {
            union { v16h v; v8h h[2]; } ua, ub;
            const _Float16* pa = A0 + (size_t)i * 16 * K_DIM + k0 + koff;
            ua.h[0] = *(const v8h*)pa;          // K = k0+koff .. +7
            ua.h[1] = *(const v8h*)(pa + 16);   // K = k0+16+koff .. +7
            a[i] = ua.v;
            const _Float16* pb = B0 + (size_t)i * 16 * K_DIM + k0 + koff;
            ub.h[0] = *(const v8h*)pb;
            ub.h[1] = *(const v8h*)(pb + 16);
            b[i] = ub.v;
        }
        #pragma unroll
        for (int i = 0; i < 4; ++i)
            #pragma unroll
            for (int j = 0; j < 4; ++j)
                acc[i][j] = __builtin_amdgcn_wmma_f32_16x16x32_f16(
                    /*neg_a=*/false, a[i], /*neg_b=*/false, b[j],
                    /*c_mod=*/(short)0, acc[i][j],
                    /*reuse_a=*/false, /*reuse_b=*/false);
    }

    // Epilogue: D layout => lane L, vgpr v : m = v + 8*(L>=16), n = L&15
    const int nn = lane & 15;
    const int mb = (lane >> 4) << 3;     // 0 or 8
    #pragma unroll
    for (int i = 0; i < 4; ++i) {
        #pragma unroll
        for (int j = 0; j < 4; ++j) {
            #pragma unroll
            for (int v = 0; v < 8; ++v) {
                int row = tm + i * 16 + mb + v;
                int col = tn + j * 16 + nn;
                float val = acc[i][j][v] * (1.0f / P_PERS);
                size_t idx = (size_t)row * N_NODES + col;
                bool keep = (mask[idx >> 5] >> (col & 31)) & 1u;
                keep = keep && (val > EPS_THR) && (row != col);
                out[idx] = keep ? val : 0.0f;
            }
        }
    }
}

// ---------------------------------------------------------------------------
extern "C" void kernel_launch(void* const* d_in, const int* in_sizes, int n_in,
                              void* d_out, int out_size, void* d_ws, size_t ws_size,
                              hipStream_t stream) {
    const float* x  = (const float*)d_in[0];   // (N, D) f32
    const float* w  = (const float*)d_in[1];   // (P, D) f32
    const int*   ei = (const int*)d_in[2];     // (2, E) i32
    float*       out = (float*)d_out;          // (N, N) f32
    const int E = in_sizes[2] / 2;

    // workspace: [ ctx f16 : 32 MB ][ edge bitmask : 8 MB ]
    _Float16* ctx = (_Float16*)d_ws;
    size_t ctx_bytes = (size_t)N_NODES * K_DIM * sizeof(_Float16);
    unsigned* mask = (unsigned*)((char*)d_ws + ctx_bytes);
    size_t mask_bytes = (size_t)N_NODES * (N_NODES / 8);

    hipMemsetAsync(mask, 0, mask_bytes, stream);
    edge_mask_kernel<<<(E + 255) / 256, 256, 0, stream>>>(ei, E, mask);
    ctx_normalize_kernel<<<dim3(P_PERS, N_NODES), D_FEAT, 0, stream>>>(x, w, ctx);
    adj_wmma_kernel<<<dim3(N_NODES / 64, N_NODES / 64), 32, 0, stream>>>(ctx, mask, out);
}
